// WindowAttentionTem_34634616275340
// MI455X (gfx1250) — matrix operand order repaired and checked by reading, not compiled, not run on hardware
//
#include <hip/hip_runtime.h>

// ---- problem dims (from reference) ----
#define BWIN 2048   // number of windows
#define NTOK 64     // tokens per window
#define CDIM 512    // channels
#define NHEAD 16
#define HDIM 32
#define NWMASK 64
#define QK_SCALE 0.17677669529663687f  // 32^-0.5

typedef __attribute__((ext_vector_type(16))) __bf16 v16bf;
typedef __attribute__((ext_vector_type(8)))  __bf16 v8bf;
typedef __attribute__((ext_vector_type(8)))  float  v8f;
typedef __attribute__((ext_vector_type(4)))  float  v4f;

static __device__ __forceinline__ v8f splat8(float x) {
  v8f r;
#pragma unroll
  for (int i = 0; i < 8; ++i) r[i] = x;
  return r;
}

static __device__ __forceinline__ v16bf mk16(v8bf lo, v8bf hi) {
  v16bf r;
#pragma unroll
  for (int i = 0; i < 8; ++i) { r[i] = lo[i]; r[i + 8] = hi[i]; }
  return r;
}

static __device__ __forceinline__ v8bf ldbf8(const __bf16* p) {
  return *(const v8bf*)p;  // single b128 load (LDS or global)
}

// A fragment (16x32 bf16, row-major rows): lane reads K run
// [aof..aof+15]; aof = (lane<16) ? 0 : 16
static __device__ __forceinline__ v16bf afrag_bf(const __bf16* row, int aof) {
  return mk16(ldbf8(row + aof), ldbf8(row + aof + 8));
}
// B fragment (32x16 bf16): lane reads 16 contiguous K values
static __device__ __forceinline__ v16bf bfrag_bf(const __bf16* p) {
  return mk16(ldbf8(p), ldbf8(p + 8));
}

static __device__ __forceinline__ v8f wmma_bf16(v16bf a, v16bf b, v8f c) {
  return __builtin_amdgcn_wmma_f32_16x16x32_bf16(false, a, false, b, (short)0, c,
                                                 false, false);
}

static __device__ __forceinline__ float rmax16(float v) {
#pragma unroll
  for (int m = 1; m < 16; m <<= 1) v = fmaxf(v, __shfl_xor(v, m, 32));
  return v;
}
static __device__ __forceinline__ float rsum16(float v) {
#pragma unroll
  for (int m = 1; m < 16; m <<= 1) v += __shfl_xor(v, m, 32);
  return v;
}

static __device__ __forceinline__ void wave_lds_fence() {
  asm volatile("s_wait_dscnt 0" ::: "memory");
}

// ---------------------------------------------------------------------------
// Kernel 0: bulk fp32 -> bf16 conversion (round-to-nearest-even via v_cvt).
// ---------------------------------------------------------------------------
__global__ __launch_bounds__(256) void cvt_f32_to_bf16(
    const float* __restrict__ in, __bf16* __restrict__ out, long n8) {
  long i = (long)blockIdx.x * blockDim.x + threadIdx.x;
  const long stride = (long)gridDim.x * blockDim.x;
  for (; i < n8; i += stride) {
    const v4f* p = (const v4f*)(in + i * 8);
    v4f a = p[0], b = p[1];
    v8bf r;
#pragma unroll
    for (int j = 0; j < 4; ++j) { r[j] = (__bf16)a[j]; r[j + 4] = (__bf16)b[j]; }
    *(v8bf*)(out + i * 8) = r;
  }
}

struct __align__(16) WaveSmem {
  __bf16 kk[64][32];  // K, row-major (kv row, d)              4 KB
  __bf16 vT[32][64];  // V transposed (d, kv row)              4 KB
  __bf16 qt[64][32];  // q (scaled), row-major, all 4 m-tiles  4 KB
  __bf16 P[16][64];   // softmax probs for current m-tile      2 KB
};

// ---------------------------------------------------------------------------
// Kernel 1: fused QKV projection + windowed attention. One block per window,
// 4 wave32 per block, each wave owns 4 heads (no cross-wave sync needed).
// All operands bf16 (pre-converted); inner loop = b128 loads + 6 wmma.
// Writes pre-projection attention output (B, 64, 512) as bf16 to scratch.
// ---------------------------------------------------------------------------
__global__ __launch_bounds__(128) void wa_fused_qkv_attn(
    const __bf16* __restrict__ xbf, const float* __restrict__ mask,
    const __bf16* __restrict__ qkvw, const float* __restrict__ qkv_b,
    const float* __restrict__ bias_table, __bf16* __restrict__ Ows) {
  const int b = blockIdx.x;
  const int lane = threadIdx.x & 31;
  const int wv = threadIdx.x >> 5;
  const int nl = lane & 15;
  const int half8 = (lane < 16) ? 0 : 8;
  const int aof = (lane < 16) ? 0 : 16;  // A-fragment K offset
  const int bof = (lane < 16) ? 0 : 16;  // B-fragment K offset
  const int wdw = b & (NWMASK - 1);

  __shared__ WaveSmem smem[4];
  WaveSmem& S = smem[wv];
  const __bf16* xb = xbf + (size_t)b * (NTOK * CDIM);

  for (int hi = 0; hi < 4; ++hi) {
    const int h = wv + 4 * hi;

    // ---------- Phase A: Q, K, V projections for this head ----------
    // 6 output col-tiles (q0,q1,k0,k1,v0,v1) share each A fragment of x.
    for (int mt = 0; mt < 4; ++mt) {
      v8f acc[6];
      int cols[6];
#pragma unroll
      for (int j = 0; j < 6; ++j) {
        cols[j] = h * HDIM + (j & 1) * 16 + (j >> 1) * CDIM + nl;
        acc[j] = splat8(qkv_b[cols[j]]);
      }
      const __bf16* arow = xb + (size_t)(mt * 16 + nl) * CDIM;
      for (int kt = 0; kt < 16; ++kt) {
        v16bf a = afrag_bf(arow + kt * 32, aof);
#pragma unroll
        for (int j = 0; j < 6; ++j) {
          v16bf bm = bfrag_bf(qkvw + (size_t)cols[j] * CDIM + kt * 32 + bof);
          acc[j] = wmma_bf16(a, bm, acc[j]);
        }
      }
      // Q: row-major, pre-scaled
#pragma unroll
      for (int j = 0; j < 2; ++j)
#pragma unroll
        for (int r = 0; r < 8; ++r)
          S.qt[mt * 16 + half8 + r][j * 16 + nl] =
              (__bf16)(acc[j][r] * QK_SCALE);
      // K: row-major (kv row, d)
#pragma unroll
      for (int j = 0; j < 2; ++j)
#pragma unroll
        for (int r = 0; r < 8; ++r)
          S.kk[mt * 16 + half8 + r][j * 16 + nl] = (__bf16)acc[2 + j][r];
      // V: transposed (d, kv row) -> packed 16B store per lane
#pragma unroll
      for (int j = 0; j < 2; ++j) {
        v8bf pk;
#pragma unroll
        for (int r = 0; r < 8; ++r) pk[r] = (__bf16)acc[4 + j][r];
        *(v8bf*)&S.vT[j * 16 + nl][mt * 16 + half8] = pk;
      }
    }
    wave_lds_fence();

    // ---------- Phase B: per m-tile S, softmax, O ----------
    for (int mt = 0; mt < 4; ++mt) {
      // S = q @ k^T  (K = hd = 32 -> single wmma per 16x16 tile)
      v16bf aq = afrag_bf(&S.qt[mt * 16 + nl][0], aof);
      v8f sc[4];
#pragma unroll
      for (int nt = 0; nt < 4; ++nt) {
        v16bf bk = bfrag_bf(&S.kk[nt * 16 + nl][bof]);
        sc[nt] = wmma_bf16(aq, bk, splat8(0.f));
      }
      // relative position bias + additive window mask
#pragma unroll
      for (int nt = 0; nt < 4; ++nt)
#pragma unroll
        for (int r = 0; r < 8; ++r) {
          int m = mt * 16 + half8 + r;
          int n = nt * 16 + nl;
          sc[nt][r] += bias_table[(m - n + (NTOK - 1)) * NHEAD + h] +
                       mask[((size_t)wdw * NTOK + m) * NTOK + n];
        }
      // row softmax over 64 keys (4 tiles x 16 lanes within half)
      float mx[8], rs[8];
#pragma unroll
      for (int r = 0; r < 8; ++r) {
        float v = fmaxf(fmaxf(sc[0][r], sc[1][r]), fmaxf(sc[2][r], sc[3][r]));
        mx[r] = rmax16(v);
      }
#pragma unroll
      for (int nt = 0; nt < 4; ++nt)
#pragma unroll
        for (int r = 0; r < 8; ++r) sc[nt][r] = __expf(sc[nt][r] - mx[r]);
#pragma unroll
      for (int r = 0; r < 8; ++r) {
        float v = sc[0][r] + sc[1][r] + sc[2][r] + sc[3][r];
        rs[r] = 1.f / rsum16(v);
      }
#pragma unroll
      for (int nt = 0; nt < 4; ++nt)
#pragma unroll
        for (int r = 0; r < 8; ++r)
          S.P[half8 + r][nt * 16 + nl] = (__bf16)(sc[nt][r] * rs[r]);
      wave_lds_fence();

      // O = P @ V  (K = 64 -> two wmma per 16x16 output tile)
#pragma unroll
      for (int dt = 0; dt < 2; ++dt) {
        v8f o = splat8(0.f);
#pragma unroll
        for (int k2 = 0; k2 < 2; ++k2) {
          v16bf ap = afrag_bf(&S.P[nl][k2 * 32], aof);
          v16bf bv = bfrag_bf(&S.vT[dt * 16 + nl][k2 * 32 + bof]);
          o = wmma_bf16(ap, bv, o);
        }
        __bf16* orow = Ows + ((size_t)b * NTOK + mt * 16 + half8) * CDIM +
                       h * HDIM + dt * 16 + nl;
#pragma unroll
        for (int r = 0; r < 8; ++r) orow[(size_t)r * CDIM] = (__bf16)o[r];
      }
      wave_lds_fence();  // P reused next m-tile
    }
    wave_lds_fence();  // before next head reuses this wave's LDS region
  }
}

// ---------------------------------------------------------------------------
// Kernel 2: output projection  out = O @ proj_w^T + proj_b  (fp32 out).
// Grid (2048, 8), 128 threads: each wave computes a 16x64 strip, K=512.
// ---------------------------------------------------------------------------
__global__ __launch_bounds__(128) void wa_proj(
    const __bf16* __restrict__ Ows, const __bf16* __restrict__ pw,
    const float* __restrict__ pb, float* __restrict__ out) {
  const int lane = threadIdx.x & 31;
  const int wv = threadIdx.x >> 5;
  const int nl = lane & 15;
  const int half8 = (lane < 16) ? 0 : 8;
  const int aof = (lane < 16) ? 0 : 16;
  const int bof = (lane < 16) ? 0 : 16;
  const size_t rbase = (size_t)blockIdx.x * 64 + wv * 16;
  const int cbase = blockIdx.y * 64;

  v8f acc[4];
#pragma unroll
  for (int nt = 0; nt < 4; ++nt) acc[nt] = splat8(pb[cbase + nt * 16 + nl]);

  const __bf16* arow = Ows + (rbase + nl) * CDIM;
  for (int kt = 0; kt < 16; ++kt) {
    v16bf a = afrag_bf(arow + kt * 32, aof);
#pragma unroll
    for (int nt = 0; nt < 4; ++nt) {
      v16bf bm = bfrag_bf(pw + (size_t)(cbase + nt * 16 + nl) * CDIM +
                          kt * 32 + bof);
      acc[nt] = wmma_bf16(a, bm, acc[nt]);
    }
  }
#pragma unroll
  for (int nt = 0; nt < 4; ++nt) {
    float* orow = out + (rbase + half8) * CDIM + cbase + nt * 16 + nl;
#pragma unroll
    for (int r = 0; r < 8; ++r) orow[(size_t)r * CDIM] = acc[nt][r];
  }
}

extern "C" void kernel_launch(void* const* d_in, const int* in_sizes, int n_in,
                              void* d_out, int out_size, void* d_ws,
                              size_t ws_size, hipStream_t stream) {
  (void)in_sizes; (void)n_in; (void)out_size; (void)ws_size;
  const float* x          = (const float*)d_in[0];
  const float* mask       = (const float*)d_in[1];
  const float* qkv_w      = (const float*)d_in[2];
  const float* qkv_b      = (const float*)d_in[3];
  const float* proj_w     = (const float*)d_in[4];
  const float* proj_b     = (const float*)d_in[5];
  const float* bias_table = (const float*)d_in[6];

  // scratch layout (bf16 regions, 16B-aligned)
  const size_t nX  = (size_t)BWIN * NTOK * CDIM;       // 67,108,864
  const size_t nQW = (size_t)3 * CDIM * CDIM;          // 786,432
  const size_t nPW = (size_t)CDIM * CDIM;              // 262,144
  __bf16* xbf    = (__bf16*)d_ws;
  __bf16* Ows    = xbf + nX;
  __bf16* qkvwbf = Ows + nX;
  __bf16* projwbf = qkvwbf + nQW;

  // Kernel 0: one-time (per launch) fp32 -> bf16 conversions
  cvt_f32_to_bf16<<<4096, 256, 0, stream>>>(x, xbf, (long)(nX / 8));
  cvt_f32_to_bf16<<<384, 256, 0, stream>>>(qkv_w, qkvwbf, (long)(nQW / 8));
  cvt_f32_to_bf16<<<128, 256, 0, stream>>>(proj_w, projwbf, (long)(nPW / 8));

  wa_fused_qkv_attn<<<BWIN, 128, 0, stream>>>(xbf, mask, qkvwbf, qkv_b,
                                              bias_table, Ows);
  wa_proj<<<dim3(BWIN, CDIM / 64), 128, 0, stream>>>(Ows, projwbf, proj_b,
                                                     (float*)d_out);
}
